// PrefrontalWorkingMemory_80882824118927
// MI455X (gfx1250) — matrix operand order repaired and checked by reading, not compiled, not run on hardware
//
#include <hip/hip_runtime.h>
#include <hip/hip_bf16.h>
#include <math.h>

// Problem constants (from reference): B=8, S=1024, D=1024, C=8, H=4, HD=256
#define BB   8
#define SS   1024
#define DD   1024
#define CC   8
#define HH   4
#define HDD  256
#define ALD  1032   // padded LDS row stride (bf16 elems) -> breaks 16-way bank conflicts

typedef __attribute__((ext_vector_type(16))) __bf16 v16bf;
typedef __attribute__((ext_vector_type(8)))  __bf16 v8bf;
typedef __attribute__((ext_vector_type(8)))  float  v8f;
typedef int v4i __attribute__((__vector_size__(4 * sizeof(int))));

// ---- CDNA5 async global->LDS copy (ASYNCcnt path), with portable fallback ----
#if defined(__AMDGCN__) && __has_builtin(__builtin_amdgcn_global_load_async_to_lds_b128) && \
    __has_builtin(__builtin_amdgcn_s_wait_asynccnt)
#define HAVE_ASYNC_LDS 1
#else
#define HAVE_ASYNC_LDS 0
#endif

__device__ __forceinline__ void ld16_to_lds(void* lds_dst, const void* g_src) {
#if HAVE_ASYNC_LDS
  // generic->AS pointer conversion via integer: low 32 bits of a generic LDS
  // pointer are the LDS byte offset (aperture rule), global generic == AS1.
  // Builtin signature (from clang diagnostic): (v4i AS1*, v4i AS3*, imm offset, imm cpol)
  __builtin_amdgcn_global_load_async_to_lds_b128(
      (__attribute__((address_space(1))) v4i*)(unsigned long long)g_src,
      (__attribute__((address_space(3))) v4i*)(unsigned int)(unsigned long long)lds_dst,
      0, 0);
#else
  *(int4*)lds_dst = *(const int4*)g_src;
#endif
}
__device__ __forceinline__ void ld16_commit() {
#if HAVE_ASYNC_LDS
  __builtin_amdgcn_s_wait_asynccnt(0);
#endif
}

__device__ __forceinline__ __bf16 f2bf(float f) {
  union { float f; unsigned u; } v; v.f = f;
  unsigned u = v.u + 0x7FFFu + ((v.u >> 16) & 1u);   // round-to-nearest-even
  unsigned short h = (unsigned short)(u >> 16);
  __bf16 r; __builtin_memcpy(&r, &h, 2); return r;
}
__device__ __forceinline__ float bf2f(__bf16 b) {
  unsigned short h; __builtin_memcpy(&h, &b, 2);
  union { unsigned u; float f; } v; v.u = ((unsigned)h) << 16; return v.f;
}

// ---------------------------------------------------------------------------
// f32 -> bf16 cast (weights + activations), one-shot prep (vectorized)
// ---------------------------------------------------------------------------
__global__ __launch_bounds__(256) void cast_bf16_kernel(const float* __restrict__ s,
                                                        __bf16* __restrict__ d, int n) {
  int i = (blockIdx.x * 256 + threadIdx.x) * 4;
  if (i + 3 < n) {
    float4 v = *(const float4*)(s + i);
    d[i + 0] = f2bf(v.x); d[i + 1] = f2bf(v.y);
    d[i + 2] = f2bf(v.z); d[i + 3] = f2bf(v.w);
  } else {
    for (int j = i; j < n; ++j) d[j] = f2bf(s[j]);
  }
}

// ---------------------------------------------------------------------------
// Big parallel GEMM: Out[M,N] = A[M,K=1024] @ W[N, Wld]^T(cols colofs..+K) + bias
// A bf16 row-major, W bf16 row-major, Out bf16. One block = one 16-row M tile,
// 8 waves x 8 N-tiles each. A tile staged in LDS via async global->LDS.
// ---------------------------------------------------------------------------
__global__ __launch_bounds__(256) void gemm16_wmma(
    const __bf16* __restrict__ A, const __bf16* __restrict__ W,
    int Wld, int colofs, const float* __restrict__ bias,
    __bf16* __restrict__ Out, int N)
{
  __shared__ __bf16 Alds[16 * ALD];
  const int tid = threadIdx.x;
  const int m0  = blockIdx.x * 16;

  // 16 rows x 1024 bf16 = 2048 16-byte chunks, 8 per thread, async -> LDS
  for (int i = tid; i < 16 * (DD / 8); i += 256) {
    int r = i >> 7, c = (i & 127) << 3;
    ld16_to_lds(&Alds[r * ALD + c], A + (size_t)(m0 + r) * DD + c);
  }
  ld16_commit();
  __syncthreads();

  const int wave = tid >> 5, lane = tid & 31;
  const int half = lane >> 4, l15 = lane & 15;
  const __bf16* ar = Alds + l15 * ALD;

#pragma unroll 1
  for (int nt = wave; nt < (N >> 4); nt += 8) {
    const int n0 = nt << 4;
    v8f acc = {};
    const __bf16* wr = W + (size_t)(n0 + l15) * Wld + colofs + half * 16;
#pragma unroll 4
    for (int k0 = 0; k0 < DD; k0 += 32) {
      if ((k0 & 127) == 0) __builtin_prefetch(wr + k0 + 256, 0, 3);  // near-cache prefetch
      v8bf alo = *(const v8bf*)(ar + k0 + half * 16);
      v8bf ahi = *(const v8bf*)(ar + k0 + half * 16 + 8);
      v16bf a = __builtin_shufflevector(alo, ahi, 0,1,2,3,4,5,6,7,8,9,10,11,12,13,14,15);
      v16bf b = *(const v16bf*)(wr + k0);
      acc = __builtin_amdgcn_wmma_f32_16x16x32_bf16(false, a, false, b, (short)0, acc, false, false);
    }
    const float bv = bias ? bias[n0 + l15] : 0.0f;
#pragma unroll
    for (int r = 0; r < 8; ++r) {
      int row = half * 8 + r;                       // C/D layout: lanes0-15 M=r, lanes16-31 M=8+r
      Out[(size_t)(m0 + row) * N + n0 + l15] = f2bf(acc[r] + bv);
    }
  }
}

// ---------------------------------------------------------------------------
// Importance MLP: imp[m] = gelu(x_m @ w1^T + b1) @ w2^T + b2   (one block/token)
// ---------------------------------------------------------------------------
__global__ __launch_bounds__(256) void importance_kernel(
    const float* __restrict__ x, const float* __restrict__ w1,
    const float* __restrict__ b1, const float* __restrict__ w2,
    const float* __restrict__ b2, float* __restrict__ imp)
{
  __shared__ float xs[DD];
  __shared__ float red[256];
  const int m = blockIdx.x, tid = threadIdx.x;
  // 1024 floats = 256 16-byte chunks, one per thread, async -> LDS
  ld16_to_lds(&xs[tid * 4], x + (size_t)m * DD + tid * 4);
  ld16_commit();
  __syncthreads();
  float acc = 0.f;
  for (int j = tid; j < DD / 2; j += 256) {
    const float* wr = w1 + (size_t)j * DD;
    float h = b1[j];
#pragma unroll 4
    for (int i = 0; i < DD; i += 4)
      h += xs[i]*wr[i] + xs[i+1]*wr[i+1] + xs[i+2]*wr[i+2] + xs[i+3]*wr[i+3];
    float g = 0.5f * h * (1.0f + erff(h * 0.70710678118f));   // exact GELU
    acc += g * w2[j];
  }
  red[tid] = acc;
  __syncthreads();
  for (int s = 128; s > 0; s >>= 1) { if (tid < s) red[tid] += red[tid + s]; __syncthreads(); }
  if (tid == 0) imp[m] = red[0] + b2[0];
}

// ---------------------------------------------------------------------------
// Wave-scope 16x1024 GEMM step used inside the scan: Osq(bf16, rows0..7 kept)
//  = Asq(16 x ALD bf16 LDS) @ W^T(+bias), scaled by mult. 32 waves, 2 tiles each.
// ---------------------------------------------------------------------------
__device__ __forceinline__ void wave_gemm_step(
    const __bf16* __restrict__ Asq, const __bf16* __restrict__ W,
    int Wld, int colofs, const float* __restrict__ bias,
    __bf16* __restrict__ Osq, float mult, int wave, int lane)
{
  const int half = lane >> 4, l15 = lane & 15;
  const __bf16* ar = Asq + l15 * ALD;
#pragma unroll 1
  for (int i = 0; i < 2; ++i) {
    const int n0 = (wave + i * 32) << 4;
    v8f acc = {};
    const __bf16* wr = W + (size_t)(n0 + l15) * Wld + colofs + half * 16;
#pragma unroll 4
    for (int k0 = 0; k0 < DD; k0 += 32) {
      v8bf alo = *(const v8bf*)(ar + k0 + half * 16);
      v8bf ahi = *(const v8bf*)(ar + k0 + half * 16 + 8);
      v16bf a = __builtin_shufflevector(alo, ahi, 0,1,2,3,4,5,6,7,8,9,10,11,12,13,14,15);
      v16bf b = *(const v16bf*)(wr + k0);
      acc = __builtin_amdgcn_wmma_f32_16x16x32_bf16(false, a, false, b, (short)0, acc, false, false);
    }
    if (half == 0) {                      // lanes 0-15 hold rows 0..7 == real batch rows
      const float bv = bias ? bias[n0 + l15] : 0.0f;
#pragma unroll
      for (int r = 0; r < 8; ++r)
        Osq[r * DD + n0 + l15] = f2bf((acc[r] + bv) * mult);
    }
  }
}

// ---------------------------------------------------------------------------
// Sequential scan over S steps. One workgroup (1024 threads / 32 waves) is the
// single sync domain (util[] couples all batches). K/V caches updated
// incrementally (only the evicted slot is re-projected each step).
// ---------------------------------------------------------------------------
__global__ __launch_bounds__(1024) void scan_kernel(
    const float* __restrict__ x,          // [B,S,D] f32
    const __bf16* __restrict__ Qall,      // [B,S,D] precomputed Q
    const __bf16* __restrict__ ogx,       // [B,S,D] x-half of output gate (incl. bias)
    const __bf16* __restrict__ igx,       // [B,S,D] x-half of input gate (incl. bias)
    const float* __restrict__ impall,     // [B,S]
    const __bf16* __restrict__ w_ao, const float* __restrict__ b_ao,
    const __bf16* __restrict__ w_og,      // [D,2D]
    const __bf16* __restrict__ w_ig,      // [D,2D]
    const __bf16* __restrict__ w_it, const float* __restrict__ b_it,
    const __bf16* __restrict__ w_ki, const float* __restrict__ b_ki,
    const __bf16* __restrict__ w_vi, const float* __restrict__ b_vi,
    const float* __restrict__ g_ln, const float* __restrict__ b_ln,
    float* __restrict__ Kc, float* __restrict__ Vc,   // [B,C,D] caches
    float* __restrict__ out)              // [B,S,D]
{
  __shared__ __bf16 Asq[16 * ALD];        // A operand (rows 8..15 stay zero)
  __shared__ __bf16 Osq[8 * DD];          // GEMM result rows 0..7
  __shared__ float  sc[BB * HH * CC];
  __shared__ float  attn[BB * HH * CC];
  __shared__ float  sstrg[BB * CC];
  __shared__ float  sutil[CC];
  __shared__ int    sworst[BB];
  __shared__ float  sdo[BB];
  __shared__ float  sred[2 * BB];
  __shared__ float  sflag;

  const int tid  = threadIdx.x;
  const int wave = tid >> 5, lane = tid & 31;
  const int d    = tid;                   // this thread's column 0..1023

  for (int i = tid; i < 16 * ALD; i += 1024) Asq[i] = f2bf(0.f);
  if (tid < BB * CC) sstrg[tid] = 0.f;
  if (tid < CC) sutil[tid] = 0.f;
  for (int i = tid; i < BB * CC * DD; i += 1024) {   // mem==0 => K/V == bias
    int dd = i & (DD - 1);
    Kc[i] = b_ki[dd];
    Vc[i] = b_vi[dd];
  }
  __syncthreads();

  const float scale = 0.0625f;            // 1/sqrt(HD)

#pragma unroll 1
  for (int t = 0; t < SS; ++t) {
    // ---- attention scores over C slots ----
    if (tid < BB * HH * CC) sc[tid] = 0.f;
    __syncthreads();
    {
      const int b = tid >> 7, h = (tid >> 5) & 3, c = (tid >> 2) & 7, p = tid & 3;
      const __bf16* qp = Qall + ((size_t)(b * SS + t) * DD + h * HDD + p * 64);
      const float*  kp = Kc   + ((size_t)(b * CC + c) * DD + h * HDD + p * 64);
      float s = 0.f;
#pragma unroll 4
      for (int i = 0; i < 64; ++i) s += bf2f(qp[i]) * kp[i];
      atomicAdd(&sc[(b * HH + h) * CC + c], s);
    }
    __syncthreads();
    // ---- masked softmax + global strength guard ----
    if (tid == 0) {
      float mx = -1e30f;
      for (int i = 0; i < BB * CC; ++i) mx = fmaxf(mx, sstrg[i]);
      sflag = (mx > 0.1f) ? 1.f : 0.f;
    }
    if (tid < BB * HH) {
      const int b = tid >> 2;
      float sv[CC]; bool any = false; float mx = -1e30f;
      for (int c = 0; c < CC; ++c) {
        bool pad = sstrg[b * CC + c] < 0.1f;
        any = any || !pad;
        float s = pad ? -1e9f : sc[tid * CC + c] * scale;
        sv[c] = s; mx = fmaxf(mx, s);
      }
      float sum = 0.f;
      for (int c = 0; c < CC; ++c) { sv[c] = __expf(sv[c] - mx); sum += sv[c]; }
      float inv = any ? 1.f / sum : 0.f;
      for (int c = 0; c < CC; ++c) attn[tid * CC + c] = sv[c] * inv;
    }
    __syncthreads();
    // ---- ctx = attn @ V  -> A operand ----
    {
      const int h = d >> 8;
#pragma unroll
      for (int b = 0; b < BB; ++b) {
        float cx = 0.f;
#pragma unroll
        for (int c = 0; c < CC; ++c)
          cx += attn[(b * HH + h) * CC + c] * Vc[(size_t)(b * CC + c) * DD + d];
        Asq[b * ALD + d] = f2bf(cx);
      }
    }
    __syncthreads();
    // ---- retrieved = ctx @ w_ao^T + b_ao, gated by strength flag ----
    wave_gemm_step(Asq, w_ao, DD, 0, b_ao, Osq, sflag, wave, lane);
    __syncthreads();
    float retr[BB], xv[BB];
#pragma unroll
    for (int b = 0; b < BB; ++b) {
      retr[b] = bf2f(Osq[b * DD + d]);
      xv[b]   = x[(size_t)(b * SS + t) * DD + d];
    }
    __syncthreads();
#pragma unroll
    for (int b = 0; b < BB; ++b) Asq[b * ALD + d] = Osq[b * DD + d];   // A <- retrieved
    __syncthreads();
    // ---- output gate: sigmoid(ogx + retrieved @ w_og[:,D:]^T) ----
    wave_gemm_step(Asq, w_og, 2 * DD, DD, nullptr, Osq, 1.f, wave, lane);
    __syncthreads();
#pragma unroll
    for (int b = 0; b < BB; ++b) {
      float g  = bf2f(ogx[(size_t)(b * SS + t) * DD + d]) + bf2f(Osq[b * DD + d]);
      float og = 1.f / (1.f + __expf(-g));
      out[(size_t)(b * SS + t) * DD + d] = xv[b] + og * retr[b];
    }
    __syncthreads();
    // ---- input gate -> u = x * in_g ----
    wave_gemm_step(Asq, w_ig, 2 * DD, DD, nullptr, Osq, 1.f, wave, lane);
    __syncthreads();
#pragma unroll
    for (int b = 0; b < BB; ++b) {
      float g  = bf2f(igx[(size_t)(b * SS + t) * DD + d]) + bf2f(Osq[b * DD + d]);
      float ig = 1.f / (1.f + __expf(-g));
      Asq[b * ALD + d] = f2bf(xv[b] * ig);
    }
    if (tid < 2 * BB) sred[tid] = 0.f;
    __syncthreads();
    // ---- store-value transform z = u @ w_it^T + b_it ----
    wave_gemm_step(Asq, w_it, DD, 0, b_it, Osq, 1.f, wave, lane);
    __syncthreads();
    // LayerNorm stats (wave reduce + LDS atomics)
#pragma unroll
    for (int b = 0; b < BB; ++b) {
      float z = bf2f(Osq[b * DD + d]);
      float s1 = z, s2 = z * z;
#pragma unroll
      for (int m = 16; m > 0; m >>= 1) {
        s1 += __shfl_xor(s1, m, 32);
        s2 += __shfl_xor(s2, m, 32);
      }
      if (lane == 0) { atomicAdd(&sred[2 * b], s1); atomicAdd(&sred[2 * b + 1], s2); }
    }
    __syncthreads();
    const float gl = g_ln[d], bl = b_ln[d];
#pragma unroll
    for (int b = 0; b < BB; ++b) {
      float mu  = sred[2 * b] * (1.f / DD);
      float var = sred[2 * b + 1] * (1.f / DD) - mu * mu;
      float z   = bf2f(Osq[b * DD + d]);
      float svv = tanhf((z - mu) * rsqrtf(var + 1e-5f) * gl + bl);
      Asq[b * ALD + d] = f2bf(svv);                 // A <- sv
    }
    // ---- eviction decision (argmin strg/(util+1), first-min tie break) ----
    if (tid < BB) {
      const int b = tid;
      int wbest = 0; float best = 3.4e38f;
      for (int c = 0; c < CC; ++c) {
        float s = sstrg[b * CC + c] / (sutil[c] + 1.f);
        if (s < best) { best = s; wbest = c; }
      }
      float cur = sstrg[b * CC + wbest];
      float im  = impall[(size_t)b * SS + t];
      float ds  = (im > cur) ? 1.f : 0.f;
      sworst[b] = wbest; sdo[b] = ds;
      if (ds > 0.f) sstrg[b * CC + wbest] = fminf(fmaxf(im, 0.f), 1.f);
    }
    __syncthreads();
    if (tid == 0) {
      for (int b = 0; b < BB; ++b) if (sdo[b] > 0.f) sutil[sworst[b]] += 1.f;
    }
    __syncthreads();
    // ---- incremental K cache refresh for stored slots ----
    wave_gemm_step(Asq, w_ki, DD, 0, b_ki, Osq, 1.f, wave, lane);
    __syncthreads();
#pragma unroll
    for (int b = 0; b < BB; ++b)
      if (sdo[b] > 0.f) Kc[(size_t)(b * CC + sworst[b]) * DD + d] = bf2f(Osq[b * DD + d]);
    __syncthreads();
    // ---- incremental V cache refresh ----
    wave_gemm_step(Asq, w_vi, DD, 0, b_vi, Osq, 1.f, wave, lane);
    __syncthreads();
#pragma unroll
    for (int b = 0; b < BB; ++b)
      if (sdo[b] > 0.f) Vc[(size_t)(b * CC + sworst[b]) * DD + d] = bf2f(Osq[b * DD + d]);
    __syncthreads();
  }
}

// ---------------------------------------------------------------------------
extern "C" void kernel_launch(void* const* d_in, const int* in_sizes, int n_in,
                              void* d_out, int out_size, void* d_ws, size_t ws_size,
                              hipStream_t stream)
{
  (void)in_sizes; (void)n_in; (void)out_size; (void)ws_size;
  const float* x    = (const float*)d_in[0];
  const float* w_qp = (const float*)d_in[1];  const float* b_qp = (const float*)d_in[2];
  const float* w_qi = (const float*)d_in[3];  const float* b_qi = (const float*)d_in[4];
  const float* w_ki = (const float*)d_in[5];  const float* b_ki = (const float*)d_in[6];
  const float* w_vi = (const float*)d_in[7];  const float* b_vi = (const float*)d_in[8];
  const float* w_ao = (const float*)d_in[9];  const float* b_ao = (const float*)d_in[10];
  const float* w_og = (const float*)d_in[11]; const float* b_og = (const float*)d_in[12];
  const float* w_ig = (const float*)d_in[13]; const float* b_ig = (const float*)d_in[14];
  const float* w_it = (const float*)d_in[15]; const float* b_it = (const float*)d_in[16];
  const float* g_ln = (const float*)d_in[17]; const float* b_ln = (const float*)d_in[18];
  const float* w_i1 = (const float*)d_in[19]; const float* b_i1 = (const float*)d_in[20];
  const float* w_i2 = (const float*)d_in[21]; const float* b_i2 = (const float*)d_in[22];
  float* out = (float*)d_out;

  char* p = (char*)d_ws;
  auto take = [&](size_t bytes) { char* r = p; p += (bytes + 255) & ~(size_t)255; return r; };
  const size_t MT = (size_t)BB * SS * DD;                 // 8M activation elements
  const size_t WD = (size_t)DD * DD;                      // 1M weight elements

  __bf16* x_bf  = (__bf16*)take(MT * 2);
  __bf16* q_bf  = (__bf16*)take(MT * 2);
  __bf16* Q_bf  = (__bf16*)take(MT * 2);
  __bf16* ogx   = (__bf16*)take(MT * 2);
  __bf16* igx   = (__bf16*)take(MT * 2);
  __bf16* wqp_b = (__bf16*)take(WD * 2);
  __bf16* wqi_b = (__bf16*)take(WD * 2);
  __bf16* wki_b = (__bf16*)take(WD * 2);
  __bf16* wvi_b = (__bf16*)take(WD * 2);
  __bf16* wao_b = (__bf16*)take(WD * 2);
  __bf16* wit_b = (__bf16*)take(WD * 2);
  __bf16* wog_b = (__bf16*)take(WD * 4);                  // [D, 2D]
  __bf16* wig_b = (__bf16*)take(WD * 4);
  float*  impw  = (float*)take((size_t)BB * SS * 4);
  float*  Kc    = (float*)take((size_t)BB * CC * DD * 4);
  float*  Vc    = (float*)take((size_t)BB * CC * DD * 4);

  // --- one-shot bf16 conversion (weights stay L2-resident afterwards) ---
  cast_bf16_kernel<<<(int)((MT/4 + 255) / 256), 256, 0, stream>>>(x,    x_bf,  (int)MT);
  cast_bf16_kernel<<<(int)((WD/4 + 255) / 256), 256, 0, stream>>>(w_qp, wqp_b, (int)WD);
  cast_bf16_kernel<<<(int)((WD/4 + 255) / 256), 256, 0, stream>>>(w_qi, wqi_b, (int)WD);
  cast_bf16_kernel<<<(int)((WD/4 + 255) / 256), 256, 0, stream>>>(w_ki, wki_b, (int)WD);
  cast_bf16_kernel<<<(int)((WD/4 + 255) / 256), 256, 0, stream>>>(w_vi, wvi_b, (int)WD);
  cast_bf16_kernel<<<(int)((WD/4 + 255) / 256), 256, 0, stream>>>(w_ao, wao_b, (int)WD);
  cast_bf16_kernel<<<(int)((WD/4 + 255) / 256), 256, 0, stream>>>(w_it, wit_b, (int)WD);
  cast_bf16_kernel<<<(int)((2*WD/4 + 255) / 256), 256, 0, stream>>>(w_og, wog_b, (int)(2*WD));
  cast_bf16_kernel<<<(int)((2*WD/4 + 255) / 256), 256, 0, stream>>>(w_ig, wig_b, (int)(2*WD));

  // --- step-independent heavy GEMMs (M = B*S = 8192), all WMMA ---
  const int MB = (int)(MT / DD / 16);                     // 512 M-tiles
  gemm16_wmma<<<MB, 256, 0, stream>>>(x_bf, wqp_b, DD,     0, b_qp, q_bf, DD);  // q
  gemm16_wmma<<<MB, 256, 0, stream>>>(q_bf, wqi_b, DD,     0, b_qi, Q_bf, DD);  // Q
  gemm16_wmma<<<MB, 256, 0, stream>>>(x_bf, wog_b, 2 * DD, 0, b_og, ogx,  DD);  // x-half of out gate
  gemm16_wmma<<<MB, 256, 0, stream>>>(x_bf, wig_b, 2 * DD, 0, b_ig, igx,  DD);  // x-half of in gate

  // --- importance MLP (x-only) ---
  importance_kernel<<<BB * SS, 256, 0, stream>>>(x, w_i1, b_i1, w_i2, b_i2, impw);

  // --- sequential scan: single workgroup, 32 waves, WMMA per-step GEMMs ---
  scan_kernel<<<1, 1024, 0, stream>>>(x, Q_bf, ogx, igx, impw,
                                      wao_b, b_ao, wog_b, wig_b,
                                      wit_b, b_it, wki_b, b_ki, wvi_b, b_vi,
                                      g_ln, b_ln, Kc, Vc, out);
}
